// GraphNet_90555090469711
// MI455X (gfx1250) — matrix operand order repaired and checked by reading, not compiled, not run on hardware
//
#include <hip/hip_runtime.h>
#include <stdint.h>

// GraphNet 3-level DAG == 2-layer MLP extracted from masked adjacency matrix.
//   H   = relu(X @ W[0:2048, 2048:10240] + bias[2048:10240])  [512 x 8192]
//   OUT =      H @ W[2048:10240, 10240:] + bias[10240:]       [512 x 1024]
// f16-split (hi/lo) x3 WMMA, f32 accumulation.
// Fast path: ALL operands pre-split into f16 hi/lo planes (weights also
// transposed to [N][K]); GEMM stages LDS purely with async-to-LDS b128 and
// the inner loop is ds_load_b128 -> v_wmma with no conversion VALU.

typedef __attribute__((ext_vector_type(16))) _Float16     v16h;
typedef __attribute__((ext_vector_type(8)))  _Float16     v8h;
typedef __attribute__((ext_vector_type(4)))  _Float16     v4h;
typedef __attribute__((ext_vector_type(8)))  float        v8f;
typedef __attribute__((ext_vector_type(4)))  float        v4f;
typedef __attribute__((ext_vector_type(4)))  unsigned int v4u;
typedef __attribute__((ext_vector_type(2)))  unsigned int v2u;

#define GN_NIN   2048
#define GN_NHID  8192
#define GN_NOUT  1024
#define GN_NTOT  11264
#define GN_BATCH 512

constexpr int BK  = 32;      // K slab (one WMMA K)
constexpr int STR = BK + 8;  // 40 halves/row (80 B): 16B-aligned, banks spread

__device__ __forceinline__ v8f wmma_f16(v16h a, v16h b, v8f c) {
    return __builtin_amdgcn_wmma_f32_16x16x32_f16(
        false, a, false, b, (short)0, c, false, false);
}

__device__ __forceinline__ unsigned int pack_split(float f) {
    _Float16 h = (_Float16)f;
    _Float16 l = (_Float16)(f - (float)h);
    return (unsigned)__builtin_bit_cast(unsigned short, h)
         | ((unsigned)__builtin_bit_cast(unsigned short, l) << 16);
}

// CDNA5 async global->LDS 16-byte copy (ASYNCcnt-tracked, no VGPR data).
__device__ __forceinline__ void async_copy_b128(const void* gaddr, void* lds_ptr) {
    unsigned lds_off = (unsigned)(uintptr_t)lds_ptr; // low 32 bits = LDS offset
    asm volatile("global_load_async_to_lds_b128 %0, %1, off"
                 :: "v"(lds_off), "v"(gaddr) : "memory");
}
__device__ __forceinline__ void wait_async0() {
    asm volatile("s_wait_asynccnt 0x0" ::: "memory");
}

__device__ __forceinline__ v16h cat8(v8h a, v8h b) {
    return __builtin_shufflevector(a, b, 0,1,2,3,4,5,6,7,8,9,10,11,12,13,14,15);
}
__device__ __forceinline__ v8h cat4(v4h a, v4h b) {
    return __builtin_shufflevector(a, b, 0,1,2,3,4,5,6,7);
}

// ---- prep: transpose+split f32 [K][N] block -> two f16 planes [N][K] ----
__global__ __launch_bounds__(256)
void transpose_split2(const float* __restrict__ src, long lds_,
                      _Float16* __restrict__ dh, _Float16* __restrict__ dl, int ldd)
{
    __shared__ unsigned int T[64 * 68];
    const int tid = threadIdx.x;
    const int n0  = blockIdx.x * 64;
    const int k0  = blockIdx.y * 64;
    #pragma unroll
    for (int i = 0; i < 4; ++i) {
        int idx = tid + i * 256;
        int r   = idx >> 4;               // k 0..63
        int c   = (idx & 15) << 2;        // n 0..60
        v4f v = *(const v4f*)(src + (long)(k0 + r) * lds_ + n0 + c);
        #pragma unroll
        for (int j = 0; j < 4; ++j)
            T[(c + j) * 68 + r] = pack_split(v[j]);
    }
    __syncthreads();
    #pragma unroll
    for (int i = 0; i < 4; ++i) {
        int idx = tid + i * 256;
        int nr  = idx >> 4;               // n 0..63
        int c   = (idx & 15) << 2;        // k 0..60
        v4u w = *(const v4u*)(T + nr * 68 + c);
        v2u hu, lu;
        hu[0] = __builtin_amdgcn_perm(w[1], w[0], 0x05040100u);
        hu[1] = __builtin_amdgcn_perm(w[3], w[2], 0x05040100u);
        lu[0] = __builtin_amdgcn_perm(w[1], w[0], 0x07060302u);
        lu[1] = __builtin_amdgcn_perm(w[3], w[2], 0x07060302u);
        long o = (long)(n0 + nr) * ldd + k0 + c;
        *(v4h*)(dh + o) = __builtin_bit_cast(v4h, hu);
        *(v4h*)(dl + o) = __builtin_bit_cast(v4h, lu);
    }
}

// ---- prep: split f32 -> two f16 planes, same layout (for X) ----
__global__ __launch_bounds__(256)
void split_plain(const float* __restrict__ src,
                 _Float16* __restrict__ dh, _Float16* __restrict__ dl)
{
    int i = (blockIdx.x * 256 + threadIdx.x) * 4;
    v4f v = *(const v4f*)(src + i);
    v4h h, l;
    #pragma unroll
    for (int j = 0; j < 4; ++j) {
        _Float16 hh = (_Float16)v[j];
        h[j] = hh;
        l[j] = (_Float16)(v[j] - (float)hh);
    }
    *(v4h*)(dh + i) = h;
    *(v4h*)(dl + i) = l;
}

// ---- fast GEMM: all operands pre-split f16 planes; LDS via async b128 ----
// Wave grid fixed 2(M) x 4(N). Tile: BM = 32*MFR rows, BN = 64*NFR cols.
template <int MFR, int NFR, bool RELU, bool SPLIT_OUT>
__global__ __launch_bounds__(256)
void gemm_split(const _Float16* __restrict__ Agh, const _Float16* __restrict__ Agl, int lda,
                const _Float16* __restrict__ Bgh, const _Float16* __restrict__ Bgl, int ldb,
                const float* __restrict__ bias,
                float* __restrict__ Cf, _Float16* __restrict__ Ch, _Float16* __restrict__ Cl,
                int ldc, int K)
{
    constexpr int BMt = 2 * 16 * MFR;
    constexpr int BNt = 4 * 16 * NFR;
    __shared__ _Float16 AhS[BMt * STR], AlS[BMt * STR];
    __shared__ _Float16 BhS[BNt * STR], BlS[BNt * STR];

    const int tid = threadIdx.x;
    const int m0  = blockIdx.y * BMt;
    const int n0  = blockIdx.x * BNt;

    const int lane  = tid & 31;
    const int wid   = tid >> 5;
    const int wm    = (wid & 1) * 16 * MFR;
    const int wn    = (wid >> 1) * 16 * NFR;
    const int lh    = lane & 15;
    const int khalf = lane >> 4;

    v8f acc[MFR][NFR] = {};

    for (int kt = 0; kt < K; kt += BK) {
        constexpr int ACH = (BMt * 4) / 256;  // 16B chunks/thread/plane for A
        constexpr int BCH = (BNt * 4) / 256;  // 16B chunks/thread/plane for B
        #pragma unroll
        for (int i = 0; i < ACH; ++i) {
            int idx = tid + i * 256;
            int r   = idx >> 2;
            int c   = (idx & 3) * 8;          // half offset within row
            long go = (long)(m0 + r) * lda + kt + c;
            async_copy_b128(Agh + go, AhS + r * STR + c);
            async_copy_b128(Agl + go, AlS + r * STR + c);
        }
        #pragma unroll
        for (int i = 0; i < BCH; ++i) {
            int idx = tid + i * 256;
            int r   = idx >> 2;
            int c   = (idx & 3) * 8;
            long go = (long)(n0 + r) * ldb + kt + c;
            async_copy_b128(Bgh + go, BhS + r * STR + c);
            async_copy_b128(Bgl + go, BlS + r * STR + c);
        }
        wait_async0();
        __syncthreads();

        v16h ah[MFR], al[MFR], bh[NFR], bl[NFR];
        #pragma unroll
        for (int mr = 0; mr < MFR; ++mr) {
            // A layout: lane<16 owns K 0..7 & 16..23; lane>=16 owns 8..15 & 24..31
            int ro = (wm + mr * 16 + lh) * STR + khalf * 8;
            ah[mr] = cat8(*(const v8h*)(AhS + ro), *(const v8h*)(AhS + ro + 16));
            al[mr] = cat8(*(const v8h*)(AlS + ro), *(const v8h*)(AlS + ro + 16));
        }
        #pragma unroll
        for (int nr = 0; nr < NFR; ++nr) {
            // B layout: lane<16 owns K 0..7 & 16..23; lane>=16 owns 8..15 & 24..31
            int ro = (wn + nr * 16 + lh) * STR + khalf * 8;
            bh[nr] = cat8(*(const v8h*)(BhS + ro), *(const v8h*)(BhS + ro + 16));
            bl[nr] = cat8(*(const v8h*)(BlS + ro), *(const v8h*)(BlS + ro + 16));
        }

        #pragma unroll
        for (int mr = 0; mr < MFR; ++mr)
            #pragma unroll
            for (int nr = 0; nr < NFR; ++nr) {
                acc[mr][nr] = wmma_f16(ah[mr], bh[nr], acc[mr][nr]);
                acc[mr][nr] = wmma_f16(ah[mr], bl[nr], acc[mr][nr]);
                acc[mr][nr] = wmma_f16(al[mr], bh[nr], acc[mr][nr]);
            }
        __syncthreads();
    }

    const int moff = khalf * 8;
    #pragma unroll
    for (int mr = 0; mr < MFR; ++mr) {
        #pragma unroll
        for (int nr = 0; nr < NFR; ++nr) {
            int gcol = n0 + wn + nr * 16 + lh;
            float bv = bias[gcol];
            #pragma unroll
            for (int i = 0; i < 8; ++i) {
                int grow = m0 + wm + mr * 16 + moff + i;
                float v = acc[mr][nr][i] + bv;
                if (RELU) v = v > 0.0f ? v : 0.0f;
                if (SPLIT_OUT) {
                    _Float16 h = (_Float16)v;
                    Ch[(long)grow * ldc + gcol] = h;
                    Cl[(long)grow * ldc + gcol] = (_Float16)(v - (float)h);
                } else {
                    Cf[(long)grow * ldc + gcol] = v;
                }
            }
        }
    }
}

// ---- fallback GEMM (small ws): in-kernel weight transform, f32 inputs ----
template <bool RELU>
__global__ __launch_bounds__(256)
void gemm_fallback(const float* __restrict__ A, int lda,
                   const float* __restrict__ B, long ldb,
                   const float* __restrict__ bias,
                   float* __restrict__ C, int ldc, int K)
{
    constexpr int BMf = 128, BNf = 128, BSTR = BK + 4;
    __shared__ _Float16     Ah[BMf * STR], Al[BMf * STR];
    __shared__ unsigned int Bs[BNf * BSTR];

    const int tid = threadIdx.x;
    const int m0 = blockIdx.y * BMf, n0 = blockIdx.x * BNf;
    const int lane = tid & 31, wid = tid >> 5;
    const int wm = (wid & 1) * 64, wn = (wid >> 1) * 32;
    const int lh = lane & 15, khalf = lane >> 4;

    v8f acc[4][2] = {};
    for (int kt = 0; kt < K; kt += BK) {
        #pragma unroll
        for (int i = 0; i < 4; ++i) {
            int idx = tid + i * 256;
            int kr = idx >> 5, nc = (idx & 31) << 2;
            v4f bv = *(const v4f*)(B + (long)(kt + kr) * ldb + n0 + nc);
            #pragma unroll
            for (int j = 0; j < 4; ++j) Bs[(nc + j) * BSTR + kr] = pack_split(bv[j]);
        }
        #pragma unroll
        for (int i = 0; i < 4; ++i) {
            int idx = tid + i * 256;
            int r = idx >> 3, c = (idx & 7) << 2;
            v4f av = *(const v4f*)(A + (long)(m0 + r) * lda + kt + c);
            v4h hv, lv;
            #pragma unroll
            for (int j = 0; j < 4; ++j) {
                _Float16 h = (_Float16)av[j];
                hv[j] = h; lv[j] = (_Float16)(av[j] - (float)h);
            }
            *(v4h*)(Ah + r * STR + c) = hv;
            *(v4h*)(Al + r * STR + c) = lv;
        }
        __syncthreads();
        v16h ah[4], al[4], bh[2], bl[2];
        #pragma unroll
        for (int mr = 0; mr < 4; ++mr) {
            int ro = (wm + mr * 16 + lh) * STR + khalf * 8;
            ah[mr] = cat8(*(const v8h*)(Ah + ro), *(const v8h*)(Ah + ro + 16));
            al[mr] = cat8(*(const v8h*)(Al + ro), *(const v8h*)(Al + ro + 16));
        }
        #pragma unroll
        for (int nr = 0; nr < 2; ++nr) {
            const unsigned int* p = Bs + (wn + nr * 16 + lh) * BSTR + khalf * 8;
            v4u q[4];
            #pragma unroll
            for (int i = 0; i < 4; ++i) q[i] = *(const v4u*)(p + 4 * i + (i >= 2 ? 8 : 0));
            v4h hq[4], lq[4];
            #pragma unroll
            for (int i = 0; i < 4; ++i) {
                v2u hu, lu;
                hu[0] = __builtin_amdgcn_perm(q[i][1], q[i][0], 0x05040100u);
                hu[1] = __builtin_amdgcn_perm(q[i][3], q[i][2], 0x05040100u);
                lu[0] = __builtin_amdgcn_perm(q[i][1], q[i][0], 0x07060302u);
                lu[1] = __builtin_amdgcn_perm(q[i][3], q[i][2], 0x07060302u);
                hq[i] = __builtin_bit_cast(v4h, hu);
                lq[i] = __builtin_bit_cast(v4h, lu);
            }
            bh[nr] = cat8(cat4(hq[0], hq[1]), cat4(hq[2], hq[3]));
            bl[nr] = cat8(cat4(lq[0], lq[1]), cat4(lq[2], lq[3]));
        }
        #pragma unroll
        for (int mr = 0; mr < 4; ++mr)
            #pragma unroll
            for (int nr = 0; nr < 2; ++nr) {
                acc[mr][nr] = wmma_f16(ah[mr], bh[nr], acc[mr][nr]);
                acc[mr][nr] = wmma_f16(ah[mr], bl[nr], acc[mr][nr]);
                acc[mr][nr] = wmma_f16(al[mr], bh[nr], acc[mr][nr]);
            }
        __syncthreads();
    }
    const int moff = khalf * 8;
    #pragma unroll
    for (int mr = 0; mr < 4; ++mr)
        #pragma unroll
        for (int nr = 0; nr < 2; ++nr) {
            int gcol = n0 + wn + nr * 16 + lh;
            float bv = bias[gcol];
            #pragma unroll
            for (int i = 0; i < 8; ++i) {
                int grow = m0 + wm + mr * 16 + moff + i;
                float v = acc[mr][nr][i] + bv;
                if (RELU) v = v > 0.0f ? v : 0.0f;
                C[(long)grow * ldc + gcol] = v;
            }
        }
}

extern "C" void kernel_launch(void* const* d_in, const int* in_sizes, int n_in,
                              void* d_out, int out_size, void* d_ws, size_t ws_size,
                              hipStream_t stream) {
    (void)in_sizes; (void)n_in; (void)out_size;
    const float* x    = (const float*)d_in[0];
    const float* W    = (const float*)d_in[1];
    const float* bias = (const float*)d_in[2];
    float* out = (float*)d_out;

    const float* W1 = W + GN_NIN;
    const float* W2 = W + (long)GN_NIN * GN_NTOT + GN_NIN + GN_NHID;

    const size_t MB = 1024 * 1024;
    dim3 blk(256);

    if (ws_size >= 116 * MB) {
        char* p = (char*)d_ws;
        _Float16* W1h = (_Float16*)(p);             // [8192][2048] 32 MB
        _Float16* W1l = (_Float16*)(p + 32 * MB);   // 32 MB
        _Float16* W2h = (_Float16*)(p + 64 * MB);   // [1024][8192] 16 MB
        _Float16* W2l = (_Float16*)(p + 80 * MB);   // 16 MB
        _Float16* Xh  = (_Float16*)(p + 96 * MB);   // [512][2048] 2 MB
        _Float16* Xl  = (_Float16*)(p + 98 * MB);   // 2 MB
        _Float16* Hh  = (_Float16*)(p + 100 * MB);  // [512][8192] 8 MB
        _Float16* Hl  = (_Float16*)(p + 108 * MB);  // 8 MB

        transpose_split2<<<dim3(GN_NHID / 64, GN_NIN / 64), blk, 0, stream>>>(
            W1, (long)GN_NTOT, W1h, W1l, GN_NIN);
        transpose_split2<<<dim3(GN_NOUT / 64, GN_NHID / 64), blk, 0, stream>>>(
            W2, (long)GN_NTOT, W2h, W2l, GN_NHID);
        split_plain<<<dim3(GN_BATCH * GN_NIN / 1024), blk, 0, stream>>>(x, Xh, Xl);

        // GEMM1: 128x128 tiles (64x4 blocks); writes H as split planes
        gemm_split<4, 2, true, true><<<dim3(GN_NHID / 128, GN_BATCH / 128), blk, 0, stream>>>(
            Xh, Xl, GN_NIN, W1h, W1l, GN_NIN, bias + GN_NIN,
            nullptr, Hh, Hl, GN_NHID, GN_NIN);
        // GEMM2: 64x64 tiles (16x8 blocks, 4x occupancy); writes f32 out
        gemm_split<2, 1, false, false><<<dim3(GN_NOUT / 64, GN_BATCH / 64), blk, 0, stream>>>(
            Hh, Hl, GN_NHID, W2h, W2l, GN_NHID, bias + GN_NIN + GN_NHID,
            out, nullptr, nullptr, GN_NOUT, GN_NHID);
    } else {
        float* H = (float*)d_ws;  // 16 MB
        gemm_fallback<true><<<dim3(GN_NHID / 128, GN_BATCH / 128), blk, 0, stream>>>(
            x, GN_NIN, W1, (long)GN_NTOT, bias + GN_NIN, H, GN_NHID, GN_NIN);
        gemm_fallback<false><<<dim3(GN_NOUT / 128, GN_BATCH / 128), blk, 0, stream>>>(
            H, GN_NHID, W2, (long)GN_NTOT, bias + GN_NIN + GN_NHID, out, GN_NOUT, GN_NHID);
    }
}
